// MutiAttention_76768245449226
// MI455X (gfx1250) — matrix operand order repaired and checked by reading, not compiled, not run on hardware
//
#include <hip/hip_runtime.h>
#include <hip/hip_bf16.h>
#include <math.h>

typedef __bf16 bf16;
typedef __attribute__((ext_vector_type(16))) __bf16 v16bf;
typedef __attribute__((ext_vector_type(8)))  __bf16 v8bf;
typedef __attribute__((ext_vector_type(8)))  float  v8f;
typedef __attribute__((ext_vector_type(4)))  unsigned int u32x4;
typedef __attribute__((ext_vector_type(8)))  int i32x8;
typedef __attribute__((ext_vector_type(4)))  int i32x4;

#define WMMA_BF16(a, b, c) \
    __builtin_amdgcn_wmma_f32_16x16x32_bf16(false, (a), false, (b), (short)0, (c), false, false)

#ifndef __has_builtin
#define __has_builtin(x) 0
#endif
#if __has_builtin(__builtin_amdgcn_tensor_load_to_lds) && \
    __has_builtin(__builtin_amdgcn_s_wait_tensorcnt)
#define HAVE_TDM 1
#else
#define HAVE_TDM 0
#endif

// ---------------------------------------------------------------------------
// Problem constants
// ---------------------------------------------------------------------------
#define BATCH      2
#define SEQ        2048
#define DIM        1024
#define NCHUNK     32          // H*B flattened head-chunks, each [2048 x 64]
#define DH         64          // head dim
#define ROWS       4096        // B*S
#define NELEM      4194304     // B*S*D
#define NB         2097152     // elements per batch for layernorm (S*D)

// LDS strides (bf16 elements) padded to dodge bank conflicts
#define KS_STRIDE  72          // 64 data + 8 pad  (= TDM pad 4 DWORDs / 32 DWORDs)
#define VS_STRIDE  40
#define PS_STRIDE  40

// ---------------------------------------------------------------------------
// fp32 -> bf16 conversion
// ---------------------------------------------------------------------------
__global__ void to_bf16_kernel(const float* __restrict__ src, bf16* __restrict__ dst, int n) {
    int i = blockIdx.x * blockDim.x + threadIdx.x;
    int stride = gridDim.x * blockDim.x;
    for (; i < n; i += stride) dst[i] = (bf16)src[i];
}

// ---------------------------------------------------------------------------
// Projection GEMM: C[4096x1024] = X[4096x1024] @ W^T, all bf16, fp32 accum.
// grid = (8 Ntiles, 64 Mtiles, 3 weights), block = 128 (4 waves).
// Each wave computes a 32x64 tile = 2x4 WMMA fragments, K-loop 32 x (K=32).
// 12 x 16B loads per 8 WMMAs per lane (L2-resident operands).
// ---------------------------------------------------------------------------
__global__ __launch_bounds__(128)
void gemm_qkv_kernel(const bf16* __restrict__ xb, const bf16* __restrict__ wb,
                     bf16* __restrict__ qkv) {
    const int z = blockIdx.z;
    const bf16* W = wb + (size_t)z * DIM * DIM;
    bf16* C = qkv + (size_t)z * ROWS * DIM;

    const int lane = threadIdx.x & 31;
    const int wid  = threadIdx.x >> 5;
    const int col16 = lane & 15;
    const int half  = lane >> 4;

    const int mbase = blockIdx.y * 64  + (wid >> 1) * 32;
    const int nbase = blockIdx.x * 128 + (wid & 1) * 64;

    v8f acc[2][4] = {};

    for (int k0 = 0; k0 < DIM; k0 += 32) {
        v16bf a[2], b[4];
        // A fragments (16x32, rows mbase+im*16..): lane row = col16, K split per half
        #pragma unroll
        for (int im = 0; im < 2; ++im) {
            const bf16* ap = xb + (size_t)(mbase + im * 16 + col16) * DIM + k0;
            v8bf lo = *(const v8bf*)(ap + half * 8);
            v8bf hi = *(const v8bf*)(ap + 16 + half * 8);
            #pragma unroll
            for (int j = 0; j < 8; ++j) { a[im][j] = lo[j]; a[im][j + 8] = hi[j]; }
        }
        // B fragments (32x16): B[k][n] = W[n][k]; K split per half like A
        #pragma unroll
        for (int jn = 0; jn < 4; ++jn) {
            const bf16* bp = W + (size_t)(nbase + jn * 16 + col16) * DIM + k0;
            v8bf lo = *(const v8bf*)(bp + half * 8);
            v8bf hi = *(const v8bf*)(bp + 16 + half * 8);
            #pragma unroll
            for (int j = 0; j < 8; ++j) { b[jn][j] = lo[j]; b[jn][j + 8] = hi[j]; }
        }
        #pragma unroll
        for (int im = 0; im < 2; ++im)
            #pragma unroll
            for (int jn = 0; jn < 4; ++jn)
                acc[im][jn] = WMMA_BF16(a[im], b[jn], acc[im][jn]);
    }

    // Store C as bf16 (C-frag layout: vgpr v -> row v + 8*half, lane -> col)
    #pragma unroll
    for (int im = 0; im < 2; ++im)
        #pragma unroll
        for (int jn = 0; jn < 4; ++jn) {
            int c = nbase + jn * 16 + col16;
            #pragma unroll
            for (int v = 0; v < 8; ++v) {
                int r = mbase + im * 16 + v + 8 * half;
                C[(size_t)r * DIM + c] = (bf16)acc[im][jn][v];
            }
        }
}

// ---------------------------------------------------------------------------
// TDM: issue an async 32x64 bf16 tile load (row stride 64) into padded LDS
// (row stride 72 bf16 via pad: 4 DWORDs after every 32 DWORDs).
// Descriptor per CDNA5 ISA section 8.3/8.4; 6-arg builtin form.
// ---------------------------------------------------------------------------
#if HAVE_TDM
__device__ __forceinline__ void tdm_load_k_tile(const bf16* gsrc, unsigned int lds_off) {
    unsigned long long ga = (unsigned long long)(uintptr_t)gsrc;
    u32x4 g0;
    g0[0] = 1u;                                       // count=1, user descriptor
    g0[1] = lds_off;                                  // lds_addr [63:32]
    g0[2] = (unsigned int)ga;                         // global_addr [95:64]
    g0[3] = (unsigned int)((ga >> 32) & 0x01FFFFFFu)  // global_addr [120:96]
          | (2u << 30);                               // type=2 (image) [127:126]
    i32x8 g1;
    g1[0] = (1 << 16)        // data_size = 2 bytes
          | (1 << 20)        // pad_enable
          | (4 << 22)        // pad_interval = 32 DWORDs
          | (3 << 25);       // pad_amount   = 4 DWORDs  -> LDS row stride 72 bf16
    g1[1] = (64 << 16);      // tensor_dim0 = 64          [79:48] low half
    g1[2] = (2048 << 16);    // tensor_dim1 = 2048        [111:80] low half
    g1[3] = (64 << 16);      // tile_dim0 = 64            [127:112]
    g1[4] = 32;              // tile_dim1 = 32            [143:128]
    g1[5] = 64;              // tensor_dim0_stride = 64   [207:160] low half
    g1[6] = 0;
    g1[7] = 0;
    i32x4 z4 = {0, 0, 0, 0};
    i32x8 z8 = {0, 0, 0, 0, 0, 0, 0, 0};
    __builtin_amdgcn_tensor_load_to_lds(g0, g1, z4, z4, z8, 0);
}
#endif

// ---------------------------------------------------------------------------
// Flash attention over one [2048 x 64] head-chunk + residual add.
// grid = (32 q-blocks, 32 chunks), block = 128 (4 waves, 16 q-rows each).
// K tiles stream LDS-ward via double-buffered TDM; V is VALU-transposed.
// ---------------------------------------------------------------------------
__global__ __launch_bounds__(128)
void attention_kernel(const bf16* __restrict__ qkv, const float* __restrict__ x,
                      float* __restrict__ y) {
    __shared__ __align__(16) bf16 Ks[2][32 * KS_STRIDE];   // K blocks [k_local][d], padded
    __shared__ __align__(16) bf16 Vt[64 * VS_STRIDE];      // V block transposed [d_v][k_local]
    __shared__ __align__(16) bf16 Ps[4][16 * PS_STRIDE];   // per-wave P staging [q][k_local]

    const int g = blockIdx.y;                  // head-chunk 0..31
    const size_t coff = (size_t)g * SEQ * DH;
    const bf16* Q = qkv + coff;
    const bf16* K = qkv + (size_t)ROWS * DIM + coff;
    const bf16* V = qkv + (size_t)2 * ROWS * DIM + coff;

    const int lane = threadIdx.x & 31;
    const int wid  = threadIdx.x >> 5;
    const int col16 = lane & 15;
    const int half  = lane >> 4;
    const int qbase = blockIdx.x * 64 + wid * 16;

    // Preload Q A-fragments for both K=32 steps of d=64
    v16bf qA[2];
    #pragma unroll
    for (int t = 0; t < 2; ++t) {
        const bf16* ap = Q + (size_t)(qbase + col16) * DH + t * 32;
        v8bf lo = *(const v8bf*)(ap + half * 8);
        v8bf hi = *(const v8bf*)(ap + 16 + half * 8);
        #pragma unroll
        for (int j = 0; j < 8; ++j) { qA[t][j] = lo[j]; qA[t][j + 8] = hi[j]; }
    }

    float m[8], l[8];
    v8f o[4] = {};
    #pragma unroll
    for (int v = 0; v < 8; ++v) { m[v] = -INFINITY; l[v] = 0.0f; }

#if HAVE_TDM
    if (wid == 0)
        tdm_load_k_tile(K, (unsigned int)(uintptr_t)&Ks[0][0]);   // prologue: tile 0
#endif

    for (int kb = 0; kb < SEQ; kb += 32) {
        const int buf = (kb >> 5) & 1;
        __syncthreads();   // prior compute done: Vt and Ks[buf^1] are free

        // Cooperative V load + transpose into LDS: 128 threads x 16 bf16
        {
            int t  = threadIdx.x;
            int kr = t >> 2;               // 0..31 (k_local)
            int cs = (t & 3) * 16;         // 0,16,32,48 (d offset)
            const bf16* vsrc = V + (size_t)(kb + kr) * DH + cs;
            #pragma unroll
            for (int j = 0; j < 16; ++j)
                Vt[(cs + j) * VS_STRIDE + kr] = vsrc[j];
#if !HAVE_TDM
            const v8bf* ksrc = (const v8bf*)(K + (size_t)(kb + kr) * DH + cs);
            *(v8bf*)(&Ks[buf][kr * KS_STRIDE + cs])     = ksrc[0];
            *(v8bf*)(&Ks[buf][kr * KS_STRIDE + cs + 8]) = ksrc[1];
#endif
        }
#if HAVE_TDM
        if (wid == 0) {
            if (kb + 32 < SEQ) {
                // prefetch next K tile into the other buffer, then wait until the
                // current tile (issued one iteration ago) has landed
                tdm_load_k_tile(K + (size_t)(kb + 32) * DH,
                                (unsigned int)(uintptr_t)&Ks[buf ^ 1][0]);
                __builtin_amdgcn_s_wait_tensorcnt(1);
            } else {
                __builtin_amdgcn_s_wait_tensorcnt(0);
            }
        }
#endif
        if (kb + 32 < SEQ)
            __builtin_prefetch(V + (size_t)(kb + 32) * DH, 0, 1);
        __syncthreads();   // Ks[buf] and Vt ready for everyone

        const bf16* KsB = &Ks[buf][0];

        // Scores: two 16x16 key tiles, each accumulated over d in two K=32 WMMAs
        v8f s[2];
        #pragma unroll
        for (int tn = 0; tn < 2; ++tn) {
            v8f c = {};
            #pragma unroll
            for (int t = 0; t < 2; ++t) {
                const bf16* bp = &KsB[(tn * 16 + col16) * KS_STRIDE + t * 32];
                v8bf lo = *(const v8bf*)(bp + half * 8);
                v8bf hi = *(const v8bf*)(bp + 16 + half * 8);
                v16bf b;
                #pragma unroll
                for (int j = 0; j < 8; ++j) { b[j] = lo[j]; b[j + 8] = hi[j]; }
                c = WMMA_BF16(qA[t], b, c);
            }
            s[tn] = c;
        }

        // Online softmax over the 32 new keys (row-wise across 16 lanes/half)
        float alpha[8];
        #pragma unroll
        for (int v = 0; v < 8; ++v) {
            float mloc = fmaxf(s[0][v], s[1][v]);
            #pragma unroll
            for (int msk = 8; msk >= 1; msk >>= 1)
                mloc = fmaxf(mloc, __shfl_xor(mloc, msk, 32));
            float mn = fmaxf(m[v], mloc);
            float a  = __expf(m[v] - mn);
            float p0 = __expf(s[0][v] - mn);
            float p1 = __expf(s[1][v] - mn);
            float ls = p0 + p1;
            #pragma unroll
            for (int msk = 8; msk >= 1; msk >>= 1)
                ls += __shfl_xor(ls, msk, 32);
            m[v] = mn;
            l[v] = l[v] * a + ls;
            alpha[v] = a;
            s[0][v] = p0;
            s[1][v] = p1;
        }
        #pragma unroll
        for (int nt = 0; nt < 4; ++nt)
            #pragma unroll
            for (int v = 0; v < 8; ++v) o[nt][v] *= alpha[v];

        // Re-layout P (C-frag) -> A-frag via per-wave LDS slab
        #pragma unroll
        for (int v = 0; v < 8; ++v) {
            int r = v + 8 * half;
            Ps[wid][r * PS_STRIDE + col16]      = (bf16)s[0][v];
            Ps[wid][r * PS_STRIDE + 16 + col16] = (bf16)s[1][v];
        }
        v16bf pA;
        {
            const bf16* pp = &Ps[wid][col16 * PS_STRIDE];
            v8bf lo = *(const v8bf*)(pp + half * 8);
            v8bf hi = *(const v8bf*)(pp + 16 + half * 8);
            #pragma unroll
            for (int j = 0; j < 8; ++j) { pA[j] = lo[j]; pA[j + 8] = hi[j]; }
        }

        // O += P x V : 4 WMMAs across the 64-wide value dim
        #pragma unroll
        for (int nt = 0; nt < 4; ++nt) {
            const bf16* bp = &Vt[(nt * 16 + col16) * VS_STRIDE];
            v8bf lo = *(const v8bf*)(bp + half * 8);
            v8bf hi = *(const v8bf*)(bp + 16 + half * 8);
            v16bf b;
            #pragma unroll
            for (int j = 0; j < 8; ++j) { b[j] = lo[j]; b[j + 8] = hi[j]; }
            o[nt] = WMMA_BF16(pA, b, o[nt]);
        }
    }

    // Finalize: y = softmax(QK^T)V + x  (flat layout == reference reshapes)
    #pragma unroll
    for (int v = 0; v < 8; ++v) {
        float invl = 1.0f / l[v];
        int r = qbase + v + 8 * half;
        #pragma unroll
        for (int nt = 0; nt < 4; ++nt) {
            int c = nt * 16 + col16;
            size_t idx = coff + (size_t)r * DH + c;
            y[idx] = o[nt][v] * invl + x[idx];
        }
    }
}

// ---------------------------------------------------------------------------
// LayerNorm over (S, D) per batch element
// ---------------------------------------------------------------------------
__global__ void zero_stats_kernel(float* stats) {
    if (threadIdx.x < 4) stats[threadIdx.x] = 0.0f;
}

__global__ __launch_bounds__(256)
void reduce_stats_kernel(const float* __restrict__ y, float* __restrict__ stats) {
    __shared__ float ssum[256], ssq[256];
    const int b = blockIdx.y;
    const size_t base = (size_t)b * NB + (size_t)blockIdx.x * 2048;
    float s = 0.0f, q = 0.0f;
    for (int i = threadIdx.x; i < 2048; i += 256) {
        float v = y[base + i];
        s += v; q += v * v;
    }
    ssum[threadIdx.x] = s; ssq[threadIdx.x] = q;
    __syncthreads();
    for (int off = 128; off > 0; off >>= 1) {
        if ((int)threadIdx.x < off) {
            ssum[threadIdx.x] += ssum[threadIdx.x + off];
            ssq[threadIdx.x]  += ssq[threadIdx.x + off];
        }
        __syncthreads();
    }
    if (threadIdx.x == 0) {
        atomicAdd(&stats[b * 2 + 0], ssum[0]);
        atomicAdd(&stats[b * 2 + 1], ssq[0]);
    }
}

__global__ __launch_bounds__(256)
void normalize_kernel(float* __restrict__ y, const float* __restrict__ stats) {
    size_t i = (size_t)blockIdx.x * blockDim.x + threadIdx.x;
    if (i >= (size_t)NELEM) return;
    int b = (int)(i >> 21);               // NB == 2^21
    const float n = (float)NB;
    float mu  = stats[b * 2 + 0] / n;
    float var = stats[b * 2 + 1] / n - mu * mu;
    float inv = rsqrtf(var + 1e-5f);
    y[i] = (y[i] - mu) * inv;
}

// ---------------------------------------------------------------------------
// Launcher
// ---------------------------------------------------------------------------
extern "C" void kernel_launch(void* const* d_in, const int* in_sizes, int n_in,
                              void* d_out, int out_size, void* d_ws, size_t ws_size,
                              hipStream_t stream) {
    const float* x  = (const float*)d_in[0];
    const float* Wq = (const float*)d_in[1];
    const float* Wk = (const float*)d_in[2];
    const float* Wv = (const float*)d_in[3];
    float* out = (float*)d_out;

    char* ws = (char*)d_ws;
    bf16* xb   = (bf16*)ws;                                   //  8 MB (x bf16)
    bf16* wb   = (bf16*)(ws + 8388608);                       //  6 MB (Wq|Wk|Wv bf16)
    bf16* qkv  = (bf16*)(ws + 8388608 + 6291456);             // 24 MB (Q|K|V bf16)
    float* stats = (float*)(ws + 8388608 + 6291456 + 25165824);

    // 1) fp32 -> bf16
    to_bf16_kernel<<<2048, 256, 0, stream>>>(x,  xb,            NELEM);
    to_bf16_kernel<<<1024, 256, 0, stream>>>(Wq, wb,            DIM * DIM);
    to_bf16_kernel<<<1024, 256, 0, stream>>>(Wk, wb + DIM * DIM,     DIM * DIM);
    to_bf16_kernel<<<1024, 256, 0, stream>>>(Wv, wb + 2 * DIM * DIM, DIM * DIM);

    // 2) Q/K/V projections via WMMA
    dim3 ggrid(DIM / 128, ROWS / 64, 3);
    gemm_qkv_kernel<<<ggrid, 128, 0, stream>>>(xb, wb, qkv);

    // 3) Flash attention per head-chunk + residual (TDM-fed K tiles)
    dim3 agrid(SEQ / 64, NCHUNK);
    attention_kernel<<<agrid, 128, 0, stream>>>(qkv, x, out);

    // 4) LayerNorm over (S, D) per batch
    zero_stats_kernel<<<1, 32, 0, stream>>>(stats);
    dim3 rgrid(NB / 2048, BATCH);
    reduce_stats_kernel<<<rgrid, 256, 0, stream>>>(out, stats);
    normalize_kernel<<<(NELEM + 255) / 256, 256, 0, stream>>>(out, stats);
}
